// MaskedConvTransformerDecoderLayer_90005334655760
// MI455X (gfx1250) — matrix operand order repaired and checked by reading, not compiled, not run on hardware
//
#include <hip/hip_runtime.h>
#include <math.h>

// Problem constants (from reference)
constexpr int B_  = 4;
constexpr int C_  = 512;
constexpr int T_  = 4096;
constexpr int H_  = 8;
constexpr int DH_ = 64;
constexpr int W_  = 64;    // half window
constexpr int NC_ = 64;    // T_/W_
constexpr int WIN3 = 192;  // 3*W_

typedef __attribute__((ext_vector_type(16))) __bf16 v16bf;
typedef __attribute__((ext_vector_type(8)))  float  v8f;

union Frag {
  v16bf v;
  unsigned short u[16];
  uint4 q[2];
};

static __device__ __forceinline__ unsigned short f2bf(float f) {
  unsigned int x = __float_as_uint(f);
  unsigned int r = x + 0x7FFFu + ((x >> 16) & 1u);   // round-to-nearest-even
  return (unsigned short)(r >> 16);
}

static __device__ __forceinline__ v8f wmma_bf16(Frag a, Frag b, v8f c) {
  // D = A(16x32 bf16) x B(32x16 bf16) + C(16x16 f32)
  return __builtin_amdgcn_wmma_f32_16x16x32_bf16(
      /*neg_a=*/false, a.v, /*neg_b=*/false, b.v,
      /*c_mod=*/(short)0, c, /*reuse_a=*/false, /*reuse_b=*/false);
}

// ---------------------------------------------------------------------------
// Stage 0a: elementwise f32 -> bf16 (weights)
// ---------------------------------------------------------------------------
__global__ void cvt_f32_bf16(const float* __restrict__ in,
                             unsigned short* __restrict__ out, int n) {
  int i = blockIdx.x * blockDim.x + threadIdx.x;
  if (i < n) out[i] = f2bf(in[i]);
}

// ---------------------------------------------------------------------------
// Stage 0b: (B, C, T) f32 -> (B, T, C) bf16 via LDS 32x32 tile transpose
// ---------------------------------------------------------------------------
__global__ __launch_bounds__(256)
void transpose_cvt(const float* __restrict__ in, unsigned short* __restrict__ out) {
  __shared__ float tile[32][33];
  const int b  = blockIdx.z;
  const int t0 = blockIdx.x * 32;
  const int c0 = blockIdx.y * 32;
  const int tx = threadIdx.x & 31;
  const int ty = threadIdx.x >> 5;   // 0..7
  const float* src = in + (size_t)b * C_ * T_;
  #pragma unroll
  for (int i = ty; i < 32; i += 8)
    tile[i][tx] = src[(size_t)(c0 + i) * T_ + t0 + tx];
  __syncthreads();
  unsigned short* dst = out + (size_t)b * T_ * C_;
  #pragma unroll
  for (int i = ty; i < 32; i += 8)
    dst[(size_t)(t0 + i) * C_ + c0 + tx] = f2bf(tile[tx][i]);
}

// ---------------------------------------------------------------------------
// Stage 1: projection GEMM.  Out[b,t,o] = (sum_c W[o,c]*x[b,c,t] + bias[o])*scale
//   A = Wbf (M=512 x K=512, row-major bf16)
//   B = XT  (per batch, N=T x K=C row-major bf16)  -> table-order K fragments (k = 8*half + s, +8 for s >= 8)
//   8 waves per WG, 32x32 tile per wave, WG tile = 64(M) x 128(N)
// ---------------------------------------------------------------------------
__global__ __launch_bounds__(256)
void gemm_bf16_proj(const unsigned short* __restrict__ Wbf,
                    const unsigned short* __restrict__ XT,
                    const float* __restrict__ bias,
                    unsigned short* __restrict__ Out,
                    float scale) {
  const int b    = blockIdx.z;
  const int m0   = blockIdx.y * 64;
  const int n0   = blockIdx.x * 128;
  const int wave = threadIdx.x >> 5;
  const int lane = threadIdx.x & 31;
  const int half = lane >> 4;
  const int l16  = lane & 15;
  const int Mb   = m0 + (wave >> 2) * 32;
  const int Nb   = n0 + (wave & 3) * 32;

  const unsigned short* Xb = XT + (size_t)b * T_ * C_;

  v8f acc[2][2] = {};
  for (int k = 0; k < C_; k += 32) {
    Frag bf[2];
    #pragma unroll
    for (int nt = 0; nt < 2; nt++) {
      const unsigned short* p =
          Xb + (size_t)(Nb + nt * 16 + l16) * C_ + k + half * 8;
      bf[nt].q[0] = *(const uint4*)(p);
      bf[nt].q[1] = *(const uint4*)(p + 16);
    }
    __builtin_prefetch(Xb + (size_t)(Nb + l16) * C_ + k + 64, 0, 1);
    #pragma unroll
    for (int mt = 0; mt < 2; mt++) {
      Frag af;
      const unsigned short* p =
          Wbf + (size_t)(Mb + mt * 16 + l16) * C_ + k + half * 8;
      af.q[0] = *(const uint4*)(p);
      af.q[1] = *(const uint4*)(p + 16);
      #pragma unroll
      for (int nt = 0; nt < 2; nt++)
        acc[mt][nt] = wmma_bf16(af, bf[nt], acc[mt][nt]);
    }
  }

  unsigned short* Ob = Out + (size_t)b * T_ * C_;
  #pragma unroll
  for (int mt = 0; mt < 2; mt++)
    #pragma unroll
    for (int nt = 0; nt < 2; nt++) {
      const int t = Nb + nt * 16 + l16;
      #pragma unroll
      for (int r = 0; r < 8; r++) {
        const int o = Mb + mt * 16 + r + 8 * half;
        Ob[(size_t)t * C_ + o] = f2bf((acc[mt][nt][r] + bias[o]) * scale);
      }
    }
}

// ---------------------------------------------------------------------------
// Stage 2: banded attention, one WG per (chunk, head, batch), 4 waves.
//   Phase 1: S(64x192) = Qc(64x64) x Kwin^T  (WMMA, wave owns 48 columns)
//   Phase 2: banded softmax per row in LDS + transpose V into LDS
//   Phase 3: out(64x64) = P(64x192) x V(192x64)  (WMMA, wave owns 16 columns)
// ---------------------------------------------------------------------------
__global__ __launch_bounds__(128)
void banded_attn(const unsigned short* __restrict__ Q,
                 const unsigned short* __restrict__ K,
                 const unsigned short* __restrict__ V,
                 const unsigned char* __restrict__ kvm,
                 unsigned short* __restrict__ AO) {
  extern __shared__ char smem[];
  float* S = (float*)smem;                                        // [64][192] f32
  unsigned short* vT = (unsigned short*)(smem + 64 * WIN3 * 4);   // [64][192] bf16 (d, y)

  const int c  = blockIdx.x;
  const int h  = blockIdx.y;
  const int b  = blockIdx.z;
  const int t0 = c * W_;       // first query row of chunk
  const int j0 = t0 - W_;      // key position at window y=0
  const int wave = threadIdx.x >> 5;
  const int lane = threadIdx.x & 31;
  const int half = lane >> 4;
  const int l16  = lane & 15;

  const size_t base = (size_t)b * T_ * C_ + h * DH_;

  // ---- Phase 1: scores ----
  {
    v8f acc[4][3] = {};
    for (int k = 0; k < DH_; k += 32) {
      Frag bf[3];
      #pragma unroll
      for (int nt = 0; nt < 3; nt++) {
        int y = wave * 48 + nt * 16 + l16;
        int j = j0 + y;
        int jc = j < 0 ? 0 : (j > T_ - 1 ? T_ - 1 : j);
        const unsigned short* p = K + base + (size_t)jc * C_ + k + half * 8;
        bf[nt].q[0] = *(const uint4*)(p);
        bf[nt].q[1] = *(const uint4*)(p + 16);
      }
      #pragma unroll
      for (int mt = 0; mt < 4; mt++) {
        Frag af;
        const unsigned short* p =
            Q + base + (size_t)(t0 + mt * 16 + l16) * C_ + k + half * 8;
        af.q[0] = *(const uint4*)(p);
        af.q[1] = *(const uint4*)(p + 16);
        #pragma unroll
        for (int nt = 0; nt < 3; nt++)
          acc[mt][nt] = wmma_bf16(af, bf[nt], acc[mt][nt]);
      }
    }
    #pragma unroll
    for (int mt = 0; mt < 4; mt++)
      #pragma unroll
      for (int nt = 0; nt < 3; nt++)
        #pragma unroll
        for (int r = 0; r < 8; r++)
          S[(mt * 16 + r + 8 * half) * WIN3 + wave * 48 + nt * 16 + l16] =
              acc[mt][nt][r];
  }
  __syncthreads();

  // ---- Phase 2a: stage V transposed (all threads) ----
  for (int idx = threadIdx.x; idx < DH_ * WIN3; idx += 128) {
    int d = idx / WIN3, y = idx - d * WIN3;
    int j = j0 + y;
    int jc = j < 0 ? 0 : (j > T_ - 1 ? T_ - 1 : j);
    vT[idx] = V[base + (size_t)jc * C_ + d];
  }

  // ---- Phase 2b: banded softmax per row (threads 0..63) ----
  if (threadIdx.x < 64) {
    const int x = threadIdx.x;
    const int t = t0 + x;
    float* row = S + x * WIN3;
    float m = -__builtin_inff();
    for (int y = x; y <= x + 2 * W_; y++) {
      int j = j0 + y;
      float s;
      if (j < 0 || j >= T_) s = -__builtin_inff();
      else s = row[y] + (kvm[(size_t)b * T_ + j] ? 0.0f : -10000.0f);
      row[y] = s;
      m = fmaxf(m, s);
    }
    float sum = 0.0f;
    for (int y = x; y <= x + 2 * W_; y++) {
      float e = __expf(row[y] - m);
      row[y] = e;
      sum += e;
    }
    const float qm = kvm[(size_t)b * T_ + t] ? 1.0f : 0.0f;
    const float f = (sum > 0.0f ? 1.0f / sum : 0.0f) * qm;
    for (int y = 0; y < WIN3; y++)
      row[y] = (y >= x && y <= x + 2 * W_) ? row[y] * f : 0.0f;
  }
  __syncthreads();

  // ---- Phase 3: out = P x V ----
  {
    v8f acc[4] = {};
    const int Nb = wave * 16;   // d tile
    for (int k = 0; k < WIN3; k += 32) {
      Frag bf;
      const unsigned short* p = vT + (Nb + l16) * WIN3 + k + half * 8;
      bf.q[0] = *(const uint4*)(p);
      bf.q[1] = *(const uint4*)(p + 16);
      #pragma unroll
      for (int mt = 0; mt < 4; mt++) {
        Frag af;
        const float* rp = S + (mt * 16 + l16) * WIN3 + k;
        #pragma unroll
        for (int e = 0; e < 8; e++) af.u[e]     = f2bf(rp[half * 8 + e]);
        #pragma unroll
        for (int e = 0; e < 8; e++) af.u[8 + e] = f2bf(rp[half * 8 + 16 + e]);
        acc[mt] = wmma_bf16(af, bf, acc[mt]);
      }
    }
    const int d = Nb + l16;
    #pragma unroll
    for (int mt = 0; mt < 4; mt++)
      #pragma unroll
      for (int r = 0; r < 8; r++) {
        const int x = mt * 16 + r + 8 * half;
        AO[base + (size_t)(t0 + x) * C_ + d] = f2bf(acc[mt][r]);
      }
  }
}

// ---------------------------------------------------------------------------
// Stage 3: output GEMM. Out[b,o,t] = (sum_c Wp[o,c]*ao[b,t,c] + bp[o]) * qmask
// ---------------------------------------------------------------------------
__global__ __launch_bounds__(256)
void gemm_bf16_out(const unsigned short* __restrict__ Wbf,
                   const unsigned short* __restrict__ XT,
                   const float* __restrict__ bias,
                   const unsigned char* __restrict__ qmask,
                   float* __restrict__ Out) {
  const int b    = blockIdx.z;
  const int m0   = blockIdx.y * 64;
  const int n0   = blockIdx.x * 128;
  const int wave = threadIdx.x >> 5;
  const int lane = threadIdx.x & 31;
  const int half = lane >> 4;
  const int l16  = lane & 15;
  const int Mb   = m0 + (wave >> 2) * 32;
  const int Nb   = n0 + (wave & 3) * 32;

  const unsigned short* Xb = XT + (size_t)b * T_ * C_;

  v8f acc[2][2] = {};
  for (int k = 0; k < C_; k += 32) {
    Frag bf[2];
    #pragma unroll
    for (int nt = 0; nt < 2; nt++) {
      const unsigned short* p =
          Xb + (size_t)(Nb + nt * 16 + l16) * C_ + k + half * 8;
      bf[nt].q[0] = *(const uint4*)(p);
      bf[nt].q[1] = *(const uint4*)(p + 16);
    }
    #pragma unroll
    for (int mt = 0; mt < 2; mt++) {
      Frag af;
      const unsigned short* p =
          Wbf + (size_t)(Mb + mt * 16 + l16) * C_ + k + half * 8;
      af.q[0] = *(const uint4*)(p);
      af.q[1] = *(const uint4*)(p + 16);
      #pragma unroll
      for (int nt = 0; nt < 2; nt++)
        acc[mt][nt] = wmma_bf16(af, bf[nt], acc[mt][nt]);
    }
  }

  float* Ob = Out + (size_t)b * C_ * T_;
  #pragma unroll
  for (int mt = 0; mt < 2; mt++)
    #pragma unroll
    for (int nt = 0; nt < 2; nt++) {
      const int t = Nb + nt * 16 + l16;
      const float mq = qmask[(size_t)b * T_ + t] ? 1.0f : 0.0f;
      #pragma unroll
      for (int r = 0; r < 8; r++) {
        const int o = Mb + mt * 16 + r + 8 * half;
        Ob[(size_t)o * T_ + t] = (acc[mt][nt][r] + bias[o]) * mq;
      }
    }
}

// ---------------------------------------------------------------------------
extern "C" void kernel_launch(void* const* d_in, const int* in_sizes, int n_in,
                              void* d_out, int out_size, void* d_ws, size_t ws_size,
                              hipStream_t stream) {
  const float* q  = (const float*)d_in[0];
  const float* k  = (const float*)d_in[1];
  const float* v  = (const float*)d_in[2];
  const unsigned char* qxm = (const unsigned char*)d_in[3];
  const unsigned char* kvm = (const unsigned char*)d_in[4];
  const float* Wq = (const float*)d_in[5];  const float* bq = (const float*)d_in[6];
  const float* Wk = (const float*)d_in[7];  const float* bk = (const float*)d_in[8];
  const float* Wv = (const float*)d_in[9];  const float* bv = (const float*)d_in[10];
  const float* Wp = (const float*)d_in[11]; const float* bp = (const float*)d_in[12];
  float* out = (float*)d_out;

  // Workspace layout (bf16 = unsigned short)
  const size_t SZ_T = (size_t)B_ * T_ * C_;          // elements per (B,T,C) tensor
  const size_t SZ_W = (size_t)C_ * C_;               // elements per weight
  unsigned short* xT  = (unsigned short*)d_ws;                   // reused; later = AO
  unsigned short* qs  = xT + SZ_T;
  unsigned short* ks  = xT + 2 * SZ_T;
  unsigned short* vs  = xT + 3 * SZ_T;
  unsigned short* wqb = xT + 4 * SZ_T;
  unsigned short* wkb = wqb + SZ_W;
  unsigned short* wvb = wkb + SZ_W;
  unsigned short* wpb = wvb + SZ_W;

  const int nW = (int)SZ_W;
  cvt_f32_bf16<<<(nW + 255) / 256, 256, 0, stream>>>(Wq, wqb, nW);
  cvt_f32_bf16<<<(nW + 255) / 256, 256, 0, stream>>>(Wk, wkb, nW);
  cvt_f32_bf16<<<(nW + 255) / 256, 256, 0, stream>>>(Wv, wvb, nW);
  cvt_f32_bf16<<<(nW + 255) / 256, 256, 0, stream>>>(Wp, wpb, nW);

  const dim3 gT(T_ / 32, C_ / 32, B_);
  const dim3 gG(T_ / 128, C_ / 64, B_);
  const float qscale = 0.125f;   // 1/sqrt(DH)

  // q projection (scaled), then k, then v — xT buffer reused serially on stream
  transpose_cvt<<<gT, 256, 0, stream>>>(q, xT);
  gemm_bf16_proj<<<gG, 256, 0, stream>>>(wqb, xT, bq, qs, qscale);
  transpose_cvt<<<gT, 256, 0, stream>>>(k, xT);
  gemm_bf16_proj<<<gG, 256, 0, stream>>>(wkb, xT, bk, ks, 1.0f);
  transpose_cvt<<<gT, 256, 0, stream>>>(v, xT);
  gemm_bf16_proj<<<gG, 256, 0, stream>>>(wvb, xT, bv, vs, 1.0f);

  // banded attention: AO aliases the (now free) xT buffer
  unsigned short* ao = xT;
  const dim3 gA(NC_, H_, B_);
  const size_t smem = (size_t)64 * WIN3 * 4 + (size_t)64 * WIN3 * 2;  // 73728 B
  banded_attn<<<gA, 128, smem, stream>>>(qs, ks, vs, kvm, ao);

  // output projection + mask
  gemm_bf16_out<<<gG, 256, 0, stream>>>(wpb, ao, bp, qxm, out);
}
